// RPN_MDN_36249523978842
// MI455X (gfx1250) — compile-verified, matrix operand order checked by reading, not run on hardware
//
#include <hip/hip_runtime.h>
#include <hip/hip_bf16.h>
#include <math.h>

#define NIMG   8
#define KMIX   5
#define RR     96000
#define NGT    32
#define PRE    2000
#define PREP   2048
#define POST   1000
#define NWORDS 64
#define IMGSZ  1280.0f
#define NMS_T  0.7f
#define SCALE_CLAMP 4.135166556742356f
#define LOG2PI 1.8378770664093453f

// d_out layout (floats): prop_b[8*1000*4] | prop_s[8*1000] | prop_u[8*1000*2] | losses[2]
#define OUT_S_OFF (NIMG*POST*4)
#define OUT_U_OFF (OUT_S_OFF + NIMG*POST)
#define OUT_L_OFF (OUT_U_OFF + NIMG*POST*2)

typedef _Float16 v16h __attribute__((ext_vector_type(16)));
typedef float    v8f  __attribute__((ext_vector_type(8)));

__device__ __forceinline__ unsigned key_of(float f) {
  unsigned u = __float_as_uint(f);
  return (u & 0x80000000u) ? ~u : (u | 0x80000000u);
}

__device__ __forceinline__ float iou_pair(float ax0,float ay0,float ax1,float ay1,
                                          float bx0,float by0,float bx1,float by1){
  float aa = (ax1-ax0)*(ay1-ay0);
  float ab = (bx1-bx0)*(by1-by0);
  float lx = fmaxf(ax0,bx0), ly = fmaxf(ay0,by0);
  float rx = fminf(ax1,bx1), ry = fminf(ay1,by1);
  float w  = fmaxf(rx-lx,0.f), h = fmaxf(ry-ly,0.f);
  float inter = w*h;
  return inter / fmaxf(aa+ab-inter, 1e-9f);
}

// Wave32 sum-reduction through the matrix pipe: v split into hi+lo f16 so the f32
// accumulator of v_wmma_f32_16x16x32_f16 reconstructs the f32 sum (B = ones).
__device__ __forceinline__ float wave_reduce_wmma(float v){
  _Float16 hi = (_Float16)v;
  _Float16 lo = (_Float16)(v - (float)hi);
  v16h a, b;
  #pragma unroll
  for (int k = 0; k < 16; ++k) { a[k] = (_Float16)0.0f; b[k] = (_Float16)1.0f; }
  a[0] = hi; a[1] = lo;
  v8f cacc = {0.f,0.f,0.f,0.f,0.f,0.f,0.f,0.f};
  cacc = __builtin_amdgcn_wmma_f32_16x16x32_f16(false, a, false, b, (short)0, cacc, false, false);
  float s = cacc[0]+cacc[1]+cacc[2]+cacc[3]+cacc[4]+cacc[5]+cacc[6]+cacc[7];
  s += __shfl_xor(s, 16);
  return s;
}

// ---------------------------------------------------------------- kernel 1
__global__ void k_stats(const float* __restrict__ anchors, const float* __restrict__ pi,
                        const float* __restrict__ mu, const float* __restrict__ sigma,
                        const float* __restrict__ deltas,
                        float* __restrict__ scores, float* __restrict__ unc,
                        float* __restrict__ boxes){
  int i = blockIdx.x*256 + threadIdx.x;
  if (i >= NIMG*RR) return;
  int n = i / RR, r = i - n*RR;
  const float* P = pi    + (size_t)n*KMIX*RR + r;
  const float* M = mu    + (size_t)n*KMIX*RR + r;
  const float* S = sigma + (size_t)n*KMIX*RR + r;
  float p[KMIX], m[KMIX], s[KMIX];
  #pragma unroll
  for (int k = 0; k < KMIX; ++k){
    __builtin_prefetch(P + (size_t)k*RR + 2048, 0, 1);   // global_prefetch_b8
    p[k] = P[(size_t)k*RR]; m[k] = M[(size_t)k*RR]; s[k] = S[(size_t)k*RR];
  }
  float mx = p[0];
  #pragma unroll
  for (int k = 1; k < KMIX; ++k) mx = fmaxf(mx, p[k]);
  float wsum = 0.f, wts[KMIX];
  #pragma unroll
  for (int k = 0; k < KMIX; ++k){ wts[k] = expf(p[k]-mx); wsum += wts[k]; }
  float inv = 1.f / wsum;
  float logit = 0.f;
  #pragma unroll
  for (int k = 0; k < KMIX; ++k){ wts[k] *= inv; logit += wts[k]*m[k]; }
  float epis = 0.f, alea = 0.f;
  #pragma unroll
  for (int k = 0; k < KMIX; ++k){
    float d = m[k]-logit;
    epis += wts[k]*d*d;
    alea += wts[k]*s[k]*s[k];
  }
  scores[i] = logit;
  unc[(size_t)i*2]   = epis;
  unc[(size_t)i*2+1] = alea;

  const float* A = anchors + (size_t)r*4;
  float aw = A[2]-A[0], ah = A[3]-A[1];
  float cx = A[0]+0.5f*aw, cy = A[1]+0.5f*ah;
  const float* D = deltas + (size_t)i*4;
  float dx = D[0], dy = D[1];
  float dw = fminf(D[2], SCALE_CLAMP), dh = fminf(D[3], SCALE_CLAMP);
  float pcx = dx*aw+cx, pcy = dy*ah+cy;
  float pw = expf(dw)*aw, ph = expf(dh)*ah;
  float* B = boxes + (size_t)i*4;
  B[0]=pcx-0.5f*pw; B[1]=pcy-0.5f*ph; B[2]=pcx+0.5f*pw; B[3]=pcy+0.5f*ph;
}

// ---------------------------------------------------------------- radix select
__global__ void k_radix_init(unsigned* state){
  int n = threadIdx.x;
  if (n < NIMG){ state[2*n] = 0u; state[2*n+1] = PRE; }
}

__global__ void k_hist(const float* __restrict__ scores, const unsigned* __restrict__ state,
                       unsigned* __restrict__ hist, int shift){
  __shared__ unsigned hsh[256];
  hsh[threadIdx.x] = 0u;
  __syncthreads();
  int i = blockIdx.x*256 + threadIdx.x;     // RR % 256 == 0 -> block is single-image
  int n = i / RR;
  unsigned key = key_of(scores[i]);
  unsigned prefix = state[2*n];
  unsigned himask = (shift >= 24) ? 0u : (0xFFFFFFFFu << (shift+8));
  if ((key & himask) == prefix) atomicAdd(&hsh[(key>>shift)&255u], 1u);
  __syncthreads();
  unsigned v = hsh[threadIdx.x];
  if (v) atomicAdd(&hist[n*256 + threadIdx.x], v);
}

__global__ void k_radix_pick(const unsigned* __restrict__ hist, unsigned* state, int shift){
  __shared__ unsigned h[256];
  int n = blockIdx.x, t = threadIdx.x;
  h[t] = hist[n*256 + t];
  __syncthreads();
  if (t == 0){
    unsigned want = state[2*n+1];
    unsigned acc = 0;
    int b;
    for (b = 255; b > 0; --b){
      if (acc + h[b] >= want) break;
      acc += h[b];
    }
    state[2*n]   = state[2*n] | ((unsigned)b << shift);
    state[2*n+1] = want - acc;
  }
}

__global__ void k_compact(const float* __restrict__ scores, const unsigned* __restrict__ state,
                          unsigned* cnt, unsigned long long* selpack){
  int i = blockIdx.x*256 + threadIdx.x;
  int n = i / RR, r = i - n*RR;
  unsigned key = key_of(scores[i]);
  if (key >= state[2*n]){
    unsigned pos = atomicAdd(&cnt[n], 1u);
    if (pos < PREP)
      selpack[(size_t)n*PREP + pos] =
        ((unsigned long long)key << 32) | (unsigned)(0xFFFFFFFFu - (unsigned)r);
  }
}

// ---------------------------------------------------------------- sort + gather
__global__ void __launch_bounds__(1024)
k_sort_gather(const unsigned long long* __restrict__ selpack,
              const float* __restrict__ scores, const float* __restrict__ boxes,
              const float* __restrict__ unc,
              float* __restrict__ tb, float* __restrict__ ts, float* __restrict__ tu){
  __shared__ unsigned long long sh[PREP];
  int n = blockIdx.x, t = threadIdx.x;
  sh[t]        = selpack[(size_t)n*PREP + t];
  sh[t + 1024] = selpack[(size_t)n*PREP + t + 1024];
  __syncthreads();
  // bitonic sort, descending on (key<<32 | ~idx)  => key desc, idx asc
  for (int k = 2; k <= PREP; k <<= 1){
    for (int j = k >> 1; j > 0; j >>= 1){
      #pragma unroll
      for (int pass = 0; pass < 2; ++pass){
        int idx = t + pass*1024;
        int ixj = idx ^ j;
        if (ixj > idx){
          unsigned long long a = sh[idx], b = sh[ixj];
          bool descSeg = ((idx & k) == 0);
          if (descSeg ? (a < b) : (a > b)){ sh[idx] = b; sh[ixj] = a; }
        }
      }
      __syncthreads();
    }
  }
  for (int s = t; s < PRE; s += 1024){
    unsigned long long e = sh[s];
    unsigned r = 0xFFFFFFFFu - (unsigned)(e & 0xFFFFFFFFull);
    size_t src = (size_t)n*RR + r;
    size_t dst = (size_t)n*PRE + s;
    ts[dst] = scores[src];
    float4 bb = *(const float4*)(boxes + src*4);
    float4 cb;
    cb.x = fminf(fmaxf(bb.x, 0.f), IMGSZ);
    cb.y = fminf(fmaxf(bb.y, 0.f), IMGSZ);
    cb.z = fminf(fmaxf(bb.z, 0.f), IMGSZ);
    cb.w = fminf(fmaxf(bb.w, 0.f), IMGSZ);
    *(float4*)(tb + dst*4) = cb;
    tu[dst*2]   = unc[src*2];
    tu[dst*2+1] = unc[src*2+1];
  }
}

// ---------------------------------------------------------------- NMS bitmatrix
__global__ void k_supmat(const float* __restrict__ tb, unsigned* __restrict__ supmat){
  long long gid  = (long long)blockIdx.x*256 + threadIdx.x;
  long long wave = gid >> 5;
  int lane = threadIdx.x & 31;
  if (wave >= (long long)NIMG*PRE*NWORDS) return;
  int n   = (int)(wave / (PRE*NWORDS));
  int rem = (int)(wave % (PRE*NWORDS));
  int i = rem / NWORDS, w = rem % NWORDS;
  int j = w*32 + lane;
  float4 bi = *(const float4*)(tb + ((size_t)n*PRE + i)*4);
  bool pred = false;
  if (j < PRE && j > i){
    float4 bj = *(const float4*)(tb + ((size_t)n*PRE + j)*4);
    pred = iou_pair(bi.x,bi.y,bi.z,bi.w, bj.x,bj.y,bj.z,bj.w) > NMS_T;
  }
  unsigned mask = (unsigned)__ballot(pred);   // wave32
  if (lane == 0) supmat[wave] = mask;
}

// ------------------------------------------------ sequential NMS + outputs
// Suppression rows are double-buffered in LDS via gfx1250 async loads:
// 16 lanes x b128 = 256 B = exactly one row; ASYNCcnt completes in order,
// so s_wait_asynccnt 0x1 after issuing row i+1 guarantees row i has landed.
__global__ void __launch_bounds__(256)
k_nms(const unsigned* __restrict__ supmat, const float* __restrict__ tb,
      const float* __restrict__ ts, const float* __restrict__ tu,
      float* __restrict__ prop_b, float* __restrict__ prop_s, float* __restrict__ prop_u){
  __shared__ unsigned keep[NWORDS];
  __shared__ unsigned mbuf[2][NWORDS];
  __shared__ unsigned totv, cv, civ;
  __shared__ unsigned sc[256];
  int n = blockIdx.x, t = threadIdx.x;
  if (t < NWORDS){
    unsigned v = 0u;
    if (t < 62) v = 0xFFFFFFFFu;
    else if (t == 62) v = 0x0000FFFFu;   // 2000 = 62*32 + 16
    keep[t] = v;
  }
  if (t == 0){ totv = 0; cv = 0; civ = 0; }

  const unsigned* M = supmat + (size_t)n*PRE*NWORDS;

  // prologue: async prefetch of row 0 (wave 0, lanes 0-15)
  if (t < 16){
    unsigned lds0 = (unsigned)(uintptr_t)(&mbuf[0][0]) + (unsigned)t*16u;
    unsigned long long ga = (unsigned long long)(uintptr_t)M + (unsigned long long)t*16ull;
    asm volatile("global_load_async_to_lds_b128 %0, %1, off" :: "v"(lds0), "v"(ga) : "memory");
  }
  __syncthreads();

  for (int i = 0; i < PRE; ++i){
    bool notlast = (i + 1) < PRE;          // uniform
    if (notlast && t < 16){
      unsigned lds1 = (unsigned)(uintptr_t)(&mbuf[(i+1)&1][0]) + (unsigned)t*16u;
      unsigned long long ga = (unsigned long long)(uintptr_t)(M + (size_t)(i+1)*NWORDS)
                            + (unsigned long long)t*16ull;
      asm volatile("global_load_async_to_lds_b128 %0, %1, off" :: "v"(lds1), "v"(ga) : "memory");
    }
    if (notlast) asm volatile("s_wait_asynccnt 0x1" ::: "memory");
    else         asm volatile("s_wait_asynccnt 0x0" ::: "memory");
    bool kept = (keep[i>>5] >> (i & 31)) & 1u;
    __syncthreads();                        // row i visible to all waves; read-before-write on keep
    if (kept && t < NWORDS) keep[t] &= ~mbuf[i & 1][t];
    __syncthreads();
  }

  // pass 1: count valid
  for (int c = 0; c < 8; ++c){
    int s = c*256 + t;
    bool valid = false;
    if (s < PRE){
      bool kb = (keep[s>>5] >> (s & 31)) & 1u;
      const float* b = tb + ((size_t)n*PRE + s)*4;
      valid = kb && (b[2] > b[0]) && (b[3] > b[1]);
    }
    unsigned bm = (unsigned)__ballot(valid);
    if ((t & 31) == 0) atomicAdd(&totv, (unsigned)__popc(bm));
  }
  __syncthreads();

  // pass 2: stable compaction (valid first, then invalid as -inf padding)
  for (int c = 0; c < 8; ++c){
    int s = c*256 + t;
    bool valid = false, inval = false;
    if (s < PRE){
      bool kb = (keep[s>>5] >> (s & 31)) & 1u;
      const float* b = tb + ((size_t)n*PRE + s)*4;
      valid = kb && (b[2] > b[0]) && (b[3] > b[1]);
      inval = !valid;
    }
    sc[t] = (valid ? 1u : 0u) | (inval ? 0x10000u : 0u);
    __syncthreads();
    for (int off = 1; off < 256; off <<= 1){
      unsigned x = (t >= off) ? sc[t-off] : 0u;
      __syncthreads();
      sc[t] += x;
      __syncthreads();
    }
    unsigned incl = sc[t], tot = sc[255];
    if (s < PRE){
      unsigned p;
      if (valid) p = cv + (incl & 0xFFFFu) - 1u;
      else       p = totv + civ + (incl >> 16) - 1u;
      if (p < POST){
        size_t op = (size_t)n*POST + p;
        size_t ip = (size_t)n*PRE + s;
        prop_s[op] = valid ? ts[ip] : -__builtin_inff();
        *(float4*)(prop_b + op*4) = *(const float4*)(tb + ip*4);
        prop_u[op*2]   = tu[ip*2];
        prop_u[op*2+1] = tu[ip*2+1];
      }
    }
    __syncthreads();
    if (t == 0){ cv += tot & 0xFFFFu; civ += tot >> 16; }
    __syncthreads();
  }
}

// ---------------------------------------------------------------- losses
__global__ void k_bestgt(const float* __restrict__ anchors, const float* __restrict__ gt,
                         unsigned* __restrict__ bg){
  __shared__ float g4[NGT*4];
  int i = blockIdx.x*256 + threadIdx.x;
  int n = i / RR, r = i - n*RR;
  if (threadIdx.x < NGT*4) g4[threadIdx.x] = gt[n*NGT*4 + threadIdx.x];
  __syncthreads();
  const float* A = anchors + (size_t)r*4;
  float a0=A[0], a1=A[1], a2=A[2], a3=A[3];
  int lane = threadIdx.x & 31;
  #pragma unroll 4
  for (int g = 0; g < NGT; ++g){
    float v = iou_pair(g4[g*4],g4[g*4+1],g4[g*4+2],g4[g*4+3], a0,a1,a2,a3);
    #pragma unroll
    for (int d = 16; d > 0; d >>= 1) v = fmaxf(v, __shfl_xor(v, d));
    if (lane == 0) atomicMax(&bg[n*NGT + g], __float_as_uint(v));  // iou >= 0
  }
}

__global__ void __launch_bounds__(1024)
k_losses(const float* __restrict__ anchors, const float* __restrict__ gt,
         const unsigned* __restrict__ bg, const float* __restrict__ pi,
         const float* __restrict__ mu, const float* __restrict__ sigma,
         const float* __restrict__ deltas, float* __restrict__ img_loss){
  __shared__ float g4[NGT*4];
  __shared__ float bgs[NGT];
  __shared__ unsigned pos_total_sh;
  __shared__ unsigned wtot[32];
  __shared__ unsigned wbase[32];
  __shared__ unsigned chtot;
  __shared__ unsigned carry_pos, carry_neg;
  __shared__ float wsum[2][32];

  int n = blockIdx.x, t = threadIdx.x;
  int wid = t >> 5, lane = t & 31;
  if (t < NGT*4) g4[t] = gt[n*NGT*4 + t];
  if (t < NGT)   bgs[t] = __uint_as_float(bg[n*NGT + t]);
  if (t == 0){ pos_total_sh = 0; carry_pos = 0; carry_neg = 0; }
  __syncthreads();

  const int CH = (RR + 1023) / 1024;

  // phase A: total positive count
  unsigned mycnt = 0;
  for (int c = 0; c < CH; ++c){
    int r = c*1024 + t;
    if (r < RR){
      const float* A = anchors + (size_t)r*4;
      float a0=A[0], a1=A[1], a2=A[2], a3=A[3];
      float best = 0.f; bool lq = false;
      #pragma unroll 4
      for (int g = 0; g < NGT; ++g){
        float v = iou_pair(g4[g*4],g4[g*4+1],g4[g*4+2],g4[g*4+3], a0,a1,a2,a3);
        if (v > best) best = v;
        lq = lq || ((v >= bgs[g] - 1e-7f) && (bgs[g] > 0.f));
      }
      if (lq || best >= 0.7f) mycnt++;
    }
  }
  #pragma unroll
  for (int d = 16; d > 0; d >>= 1) mycnt += __shfl_xor(mycnt, d);
  if (lane == 0) atomicAdd(&pos_total_sh, mycnt);
  __syncthreads();
  unsigned n_pos = min(pos_total_sh, 128u);

  // phase B: sampling via ballot-based block cumsum + loss accumulation
  float accC = 0.f, accL = 0.f;
  for (int c = 0; c < CH; ++c){
    int r = c*1024 + t;
    bool pos = false, neg = false;
    float best = 0.f; int idx = 0;
    float a0=0.f, a1=0.f, a2=1.f, a3=1.f;
    if (r < RR){
      const float* A = anchors + (size_t)r*4;
      a0=A[0]; a1=A[1]; a2=A[2]; a3=A[3];
      bool lq = false;
      #pragma unroll 4
      for (int g = 0; g < NGT; ++g){
        float v = iou_pair(g4[g*4],g4[g*4+1],g4[g*4+2],g4[g*4+3], a0,a1,a2,a3);
        if (v > best){ best = v; idx = g; }
        lq = lq || ((v >= bgs[g] - 1e-7f) && (bgs[g] > 0.f));
      }
      pos = lq || (best >= 0.7f);
      neg = (!lq) && (best < 0.3f);
    }
    // intra-wave inclusive prefix via ballot+popc (wave32)
    unsigned bpos = (unsigned)__ballot(pos);
    unsigned bneg = (unsigned)__ballot(neg);
    unsigned below = 0xFFFFFFFFu >> (31 - lane);        // bits [0..lane]
    unsigned inclp = (unsigned)__popc(bpos & below);
    unsigned incln = (unsigned)__popc(bneg & below);
    if (lane == 0)
      wtot[wid] = (unsigned)__popc(bpos) | ((unsigned)__popc(bneg) << 16);
    __syncthreads();
    if (t < 32){
      unsigned v = wtot[t], orig = v;
      #pragma unroll
      for (int d = 1; d < 32; d <<= 1){
        unsigned x = __shfl_up(v, (unsigned)d);
        if (lane >= d) v += x;
      }
      wbase[t] = v - orig;                              // exclusive prefix of wave totals
      if (t == 31) chtot = v;                           // packed chunk totals
    }
    __syncthreads();
    unsigned base = wbase[wid];
    bool keep_pos = pos && (carry_pos + (base & 0xFFFFu) + inclp <= n_pos);
    bool keep_neg = neg && (carry_neg + (base >> 16)     + incln <= 256u - n_pos);

    if (r < RR && keep_pos){
      float sw = a2-a0, sh_ = a3-a1;
      float scx = a0+0.5f*sw, scy = a1+0.5f*sh_;
      float t0=g4[idx*4], t1=g4[idx*4+1], t2=g4[idx*4+2], t3=g4[idx*4+3];
      float tw = t2-t0, th = t3-t1, tcx = t0+0.5f*tw, tcy = t1+0.5f*th;
      float gd0 = (tcx-scx)/sw, gd1 = (tcy-scy)/sh_;
      float gd2 = logf(tw/sw),  gd3 = logf(th/sh_);
      const float* D = deltas + ((size_t)n*RR + r)*4;
      accL += fabsf(D[0]-gd0) + fabsf(D[1]-gd1) + fabsf(D[2]-gd2) + fabsf(D[3]-gd3);
    }
    if (r < RR && (keep_pos || keep_neg)){
      float tclip = fminf(fmaxf(best, 0.f), 1.f);
      const float* P = pi    + (size_t)n*KMIX*RR + r;
      const float* M = mu    + (size_t)n*KMIX*RR + r;
      const float* S = sigma + (size_t)n*KMIX*RR + r;
      float p[KMIX], m[KMIX], s[KMIX];
      #pragma unroll
      for (int k = 0; k < KMIX; ++k){
        p[k] = P[(size_t)k*RR]; m[k] = M[(size_t)k*RR]; s[k] = S[(size_t)k*RR];
      }
      float pm = p[0];
      #pragma unroll
      for (int k = 1; k < KMIX; ++k) pm = fmaxf(pm, p[k]);
      float se = 0.f;
      #pragma unroll
      for (int k = 0; k < KMIX; ++k) se += expf(p[k]-pm);
      float lse_pi = pm + logf(se);
      float comp[KMIX], cm = -__builtin_inff();
      #pragma unroll
      for (int k = 0; k < KMIX; ++k){
        float z = (tclip - m[k]) / s[k];
        comp[k] = (p[k]-lse_pi) - 0.5f*z*z - logf(s[k]) - 0.5f*LOG2PI;
        cm = fmaxf(cm, comp[k]);
      }
      float s2 = 0.f;
      #pragma unroll
      for (int k = 0; k < KMIX; ++k) s2 += expf(comp[k]-cm);
      accC += -(cm + logf(s2));
    }
    __syncthreads();
    if (t == 0){ carry_pos += chtot & 0xFFFFu; carry_neg += chtot >> 16; }
    __syncthreads();
  }

  // block reduction through WMMA (wave32 matrix pipe)
  float c_w = wave_reduce_wmma(accC);
  float l_w = wave_reduce_wmma(accL);
  if (lane == 0){ wsum[0][wid] = c_w; wsum[1][wid] = l_w; }
  __syncthreads();
  if (wid == 0){
    float c2 = wave_reduce_wmma(wsum[0][lane]);
    float l2 = wave_reduce_wmma(wsum[1][lane]);
    if (lane == 0){ img_loss[n*2] = c2; img_loss[n*2+1] = l2; }
  }
}

__global__ void k_final(const float* __restrict__ img_loss, float* __restrict__ out_losses){
  if (blockIdx.x == 0 && threadIdx.x == 0){
    float c = 0.f, l = 0.f;
    for (int n = 0; n < NIMG; ++n){ c += img_loss[n*2]; l += img_loss[n*2+1]; }
    out_losses[0] = c / 2048.0f;   // BATCH_PER_IMG * NIMG
    out_losses[1] = l / 2048.0f;
  }
}

// ---------------------------------------------------------------- launcher
extern "C" void kernel_launch(void* const* d_in, const int* in_sizes, int n_in,
                              void* d_out, int out_size, void* d_ws, size_t ws_size,
                              hipStream_t stream){
  (void)in_sizes; (void)n_in; (void)out_size; (void)ws_size;
  const float* anchors = (const float*)d_in[0];
  const float* pi      = (const float*)d_in[1];
  const float* mu      = (const float*)d_in[2];
  const float* sigma   = (const float*)d_in[3];
  const float* deltas  = (const float*)d_in[4];
  const float* gt      = (const float*)d_in[5];
  float* out = (float*)d_out;

  char* w = (char*)d_ws;
  size_t off = 0;
  auto carve = [&](size_t bytes) -> void* {
    void* p = w + off;
    off += (bytes + 255) & ~(size_t)255;
    return p;
  };
  float*    scores  = (float*)   carve((size_t)NIMG*RR*4);
  float*    unc     = (float*)   carve((size_t)NIMG*RR*8);
  float*    boxes   = (float*)   carve((size_t)NIMG*RR*16);
  float*    tb      = (float*)   carve((size_t)NIMG*PRE*16);
  float*    ts      = (float*)   carve((size_t)NIMG*PRE*4);
  float*    tu      = (float*)   carve((size_t)NIMG*PRE*8);
  unsigned long long* selpack = (unsigned long long*)carve((size_t)NIMG*PREP*8);
  unsigned* hist    = (unsigned*)carve((size_t)NIMG*256*4);
  unsigned* state   = (unsigned*)carve((size_t)NIMG*2*4);
  unsigned* cnt     = (unsigned*)carve((size_t)NIMG*4);
  unsigned* supmat  = (unsigned*)carve((size_t)NIMG*PRE*NWORDS*4);
  unsigned* bg      = (unsigned*)carve((size_t)NIMG*NGT*4);
  float*    img_loss= (float*)   carve((size_t)NIMG*2*4);

  const int nb = (NIMG*RR) / 256;   // 3000, exact

  k_stats<<<nb, 256, 0, stream>>>(anchors, pi, mu, sigma, deltas, scores, unc, boxes);

  k_radix_init<<<1, 32, 0, stream>>>(state);
  for (int p = 0; p < 4; ++p){
    int shift = 24 - 8*p;
    hipMemsetAsync(hist, 0, (size_t)NIMG*256*4, stream);
    k_hist<<<nb, 256, 0, stream>>>(scores, state, hist, shift);
    k_radix_pick<<<NIMG, 256, 0, stream>>>(hist, state, shift);
  }
  hipMemsetAsync(cnt, 0, (size_t)NIMG*4, stream);
  hipMemsetAsync(selpack, 0, (size_t)NIMG*PREP*8, stream);
  k_compact<<<nb, 256, 0, stream>>>(scores, state, cnt, selpack);
  k_sort_gather<<<NIMG, 1024, 0, stream>>>(selpack, scores, boxes, unc, tb, ts, tu);

  long long lanes = (long long)NIMG*PRE*NWORDS*32;
  k_supmat<<<(int)(lanes/256), 256, 0, stream>>>(tb, supmat);
  k_nms<<<NIMG, 256, 0, stream>>>(supmat, tb, ts, tu,
                                  out, out + OUT_S_OFF, out + OUT_U_OFF);

  hipMemsetAsync(bg, 0, (size_t)NIMG*NGT*4, stream);
  k_bestgt<<<nb, 256, 0, stream>>>(anchors, gt, bg);
  k_losses<<<NIMG, 1024, 0, stream>>>(anchors, gt, bg, pi, mu, sigma, deltas, img_loss);
  k_final<<<1, 32, 0, stream>>>(img_loss, out + OUT_L_OFF);
}
